// GCN_75660143886886
// MI455X (gfx1250) — matrix operand order repaired and checked by reading, not compiled, not run on hardware
//
#include <hip/hip_runtime.h>
#include <hip/hip_bf16.h>

#define D_CONV 128
#define D_HID  256
#define EPS    1e-5f

typedef __attribute__((ext_vector_type(16))) __bf16 v16bf;
typedef __attribute__((ext_vector_type(8)))  __bf16 v8bf;
typedef __attribute__((ext_vector_type(4)))  __bf16 v4bf;
typedef __attribute__((ext_vector_type(8)))  float  v8f;

// ---------------------------------------------------------------------------
// GEMM: XW[n][c] = sum_k X[n][k] * W[k][c]   (N x 128) @ (128 x 128), bf16 WMMA
// Block: 256 threads (8 waves), 64 rows of X, full W (transposed) in LDS.
// ---------------------------------------------------------------------------
#define TILE_M 64
#define LDSP   144   // padded row length (288B stride: multiple of 32B)

__global__ __launch_bounds__(256)
void gemm_xw(const float* __restrict__ X, const float* __restrict__ W,
             float* __restrict__ XW, int n_rows)
{
    __shared__ __bf16 sA[TILE_M][LDSP];   // sA[r][k]
    __shared__ __bf16 sW[D_CONV][LDSP];   // sW[c][k] = W[k][c] (transposed)

    const int t    = threadIdx.x;
    const int row0 = blockIdx.x * TILE_M;
    const bool full = (row0 + TILE_M) <= n_rows;   // block-uniform

    // Load W (128x128 f32) transposed into LDS as bf16: float4 global loads
    {
        const float4* W4 = (const float4*)W;
        #pragma unroll
        for (int i = 0; i < 16; ++i) {
            int idx = t + i * 256;           // float4 index (4096 total)
            int k   = idx >> 5;              // 32 float4 per W row
            int c0  = (idx & 31) * 4;
            float4 w = W4[idx];
            sW[c0 + 0][k] = (__bf16)w.x;
            sW[c0 + 1][k] = (__bf16)w.y;
            sW[c0 + 2][k] = (__bf16)w.z;
            sW[c0 + 3][k] = (__bf16)w.w;
        }
    }
    // Load A tile (TILE_M x 128 f32): float4 loads, packed 8B LDS stores
    {
        #pragma unroll
        for (int i = 0; i < (TILE_M * 32) / 256; ++i) {
            int idx = t + i * 256;           // float4 index
            int r   = idx >> 5;
            int k0  = (idx & 31) * 4;
            float4 v = {0.0f, 0.0f, 0.0f, 0.0f};
            if (full || (row0 + r) < n_rows)
                v = *(const float4*)(X + (size_t)(row0 + r) * D_CONV + k0);
            v4bf p;
            p[0] = (__bf16)v.x; p[1] = (__bf16)v.y;
            p[2] = (__bf16)v.z; p[3] = (__bf16)v.w;
            *(v4bf*)&sA[r][k0] = p;
        }
    }
    __syncthreads();

    const int wave = t >> 5, lane = t & 31;
    const int n0 = wave * 16;          // this wave's column tile
    const int m  = lane & 15;          // row-in-tile (A) / col-in-tile (B,C)
    const int hi = lane >> 4;          // lane half

    // Hoist B fragments (mt-invariant): 4 k-chunks x 16 bf16 per lane
    v16bf bfrag[4];
    #pragma unroll
    for (int kc = 0; kc < 4; ++kc)
        bfrag[kc] = *(const v16bf*)&sW[n0 + m][kc * 32 + hi * 16];

    #pragma unroll
    for (int mt = 0; mt < TILE_M / 16; ++mt) {
        const int mb = mt * 16;
        v8f acc = {};
        #pragma unroll
        for (int kc = 0; kc < 4; ++kc) {
            const int k0 = kc * 32;
            // A fragment (16x32 bf16): half hi holds K = k0+hi*16+{0..15}
            v8bf alo = *(const v8bf*)&sA[mb + m][k0 + hi * 16];
            v8bf ahi = *(const v8bf*)&sA[mb + m][k0 + hi * 16 + 8];
            v16bf a;
            #pragma unroll
            for (int j = 0; j < 8; ++j) { a[j] = alo[j]; a[j + 8] = ahi[j]; }
            acc = __builtin_amdgcn_wmma_f32_16x16x32_bf16(
                      false, a, false, bfrag[kc], (short)0, acc, false, false);
        }
        // C layout: acc[j] = C[mb + hi*8 + j][n0 + m]
        const int rbase = row0 + mb + hi * 8;
        float* outp = XW + (size_t)rbase * D_CONV + n0 + m;
        if (full) {
            #pragma unroll
            for (int j = 0; j < 8; ++j) outp[(size_t)j * D_CONV] = acc[j];
        } else {
            #pragma unroll
            for (int j = 0; j < 8; ++j)
                if (rbase + j < n_rows) outp[(size_t)j * D_CONV] = acc[j];
        }
    }
}

// ---------------------------------------------------------------------------
// CSR build (once per launch): in-degree histogram, 2-level scan, fill
// ---------------------------------------------------------------------------
__global__ void count_in(const int* __restrict__ dst, int* __restrict__ cnt, int nE)
{
    int e = blockIdx.x * blockDim.x + threadIdx.x;
    if (e < nE) atomicAdd(&cnt[dst[e]], 1);
}

__global__ __launch_bounds__(1024)
void scan_block(const int* __restrict__ cnt, int* __restrict__ incl,
                int* __restrict__ bsum, int n)
{
    __shared__ int s[1024];
    int i = blockIdx.x * 1024 + threadIdx.x;
    s[threadIdx.x] = (i < n) ? cnt[i] : 0;
    __syncthreads();
    for (int off = 1; off < 1024; off <<= 1) {
        int x = (threadIdx.x >= off) ? s[threadIdx.x - off] : 0;
        __syncthreads();
        s[threadIdx.x] += x;
        __syncthreads();
    }
    if (i < n) incl[i] = s[threadIdx.x];
    if (threadIdx.x == 1023) bsum[blockIdx.x] = s[1023];
}

__global__ void scan_sums(int* bsum, int nb)
{
    if (threadIdx.x == 0 && blockIdx.x == 0) {
        int acc = 0;
        for (int i = 0; i < nb; ++i) { int v = bsum[i]; bsum[i] = acc; acc += v; }
    }
}

__global__ void make_rowptr(const int* __restrict__ cnt, const int* __restrict__ incl,
                            const int* __restrict__ bsum, int* __restrict__ rp,
                            int n, int nE)
{
    int i = blockIdx.x * blockDim.x + threadIdx.x;
    if (i < n) rp[i] = bsum[i >> 10] + incl[i] - cnt[i];
    if (i == 0) rp[n] = nE;
}

__global__ void fill_csr(const int* __restrict__ src, const int* __restrict__ dst,
                         const int* __restrict__ rp, int* __restrict__ fill,
                         int* __restrict__ col, int nE)
{
    int e = blockIdx.x * blockDim.x + threadIdx.x;
    if (e < nE) {
        int d   = dst[e];
        int pos = atomicAdd(&fill[d], 1);
        col[rp[d] + pos] = src[e];
    }
}

__global__ void calc_dinv(const int* __restrict__ cnt, float* __restrict__ dinv, int n)
{
    int i = blockIdx.x * blockDim.x + threadIdx.x;
    if (i < n) dinv[i] = rsqrtf((float)(cnt[i] + 1));   // deg includes self loop, >= 1
}

// ---------------------------------------------------------------------------
// Aggregate: y[n] = relu( dinv[n]^2 * xw[n] + sum_{s->n} dinv[s]*dinv[n]*xw[s] + b )
// One wave per node, 4 channels per lane; BN stats via LDS then global atomics.
// ---------------------------------------------------------------------------
__global__ __launch_bounds__(256)
void aggregate(const float* __restrict__ xw, const int* __restrict__ rp,
               const int* __restrict__ col, const float* __restrict__ dinv,
               const float* __restrict__ bias, float* __restrict__ y,
               float* __restrict__ ssum, float* __restrict__ ssq, int n)
{
    __shared__ float ls[256];            // [0..127]=sum, [128..255]=sumsq
    ls[threadIdx.x] = 0.0f;
    __syncthreads();

    const int wave = threadIdx.x >> 5, lane = threadIdx.x & 31;
    const int node = blockIdx.x * 8 + wave;

    if (node < n) {
        const int c = lane * 4;
        const float dn = dinv[node];
        float4 v = *(const float4*)(xw + (size_t)node * D_CONV + c);
        float sl = dn * dn;
        float4 acc;
        acc.x = v.x * sl; acc.y = v.y * sl; acc.z = v.z * sl; acc.w = v.w * sl;

        const int jb = rp[node], je = rp[node + 1];
        for (int j = jb; j < je; ++j) {
            if (j + 8 < je) {
                int sp = col[j + 8];     // prefetch row 8 edges ahead
                __builtin_prefetch(xw + (size_t)sp * D_CONV + c, 0, 0);
            }
            int s = col[j];
            float w = dinv[s] * dn;
            float4 u = *(const float4*)(xw + (size_t)s * D_CONV + c);
            acc.x += u.x * w; acc.y += u.y * w; acc.z += u.z * w; acc.w += u.w * w;
        }
        acc.x = fmaxf(acc.x + bias[c + 0], 0.0f);
        acc.y = fmaxf(acc.y + bias[c + 1], 0.0f);
        acc.z = fmaxf(acc.z + bias[c + 2], 0.0f);
        acc.w = fmaxf(acc.w + bias[c + 3], 0.0f);
        *(float4*)(y + (size_t)node * D_CONV + c) = acc;

        atomicAdd(&ls[c + 0], acc.x);          atomicAdd(&ls[128 + c + 0], acc.x * acc.x);
        atomicAdd(&ls[c + 1], acc.y);          atomicAdd(&ls[128 + c + 1], acc.y * acc.y);
        atomicAdd(&ls[c + 2], acc.z);          atomicAdd(&ls[128 + c + 2], acc.z * acc.z);
        atomicAdd(&ls[c + 3], acc.w);          atomicAdd(&ls[128 + c + 3], acc.w * acc.w);
    }
    __syncthreads();
    float v = ls[threadIdx.x];
    if (threadIdx.x < 128) atomicAdd(&ssum[threadIdx.x], v);
    else                   atomicAdd(&ssq[threadIdx.x - 128], v);
}

// ---------------------------------------------------------------------------
// BatchNorm apply (training-mode biased stats) over node features
// ---------------------------------------------------------------------------
__global__ void bn_apply(const float* __restrict__ y, const float* __restrict__ ssum,
                         const float* __restrict__ ssq, const float* __restrict__ gamma,
                         const float* __restrict__ beta, float* __restrict__ xo, int n)
{
    int i = blockIdx.x * blockDim.x + threadIdx.x;   // over n*32 float4 chunks
    if (i >= n * 32) return;
    int c = (i & 31) * 4;
    float4 v = ((const float4*)y)[i];
    float invn = 1.0f / (float)n;
    float4 o;
    {
        float m = ssum[c + 0] * invn; float var = ssq[c + 0] * invn - m * m;
        o.x = gamma[c + 0] * (v.x - m) * rsqrtf(var + EPS) + beta[c + 0];
    }
    {
        float m = ssum[c + 1] * invn; float var = ssq[c + 1] * invn - m * m;
        o.y = gamma[c + 1] * (v.y - m) * rsqrtf(var + EPS) + beta[c + 1];
    }
    {
        float m = ssum[c + 2] * invn; float var = ssq[c + 2] * invn - m * m;
        o.z = gamma[c + 2] * (v.z - m) * rsqrtf(var + EPS) + beta[c + 2];
    }
    {
        float m = ssum[c + 3] * invn; float var = ssq[c + 3] * invn - m * m;
        o.w = gamma[c + 3] * (v.w - m) * rsqrtf(var + EPS) + beta[c + 3];
    }
    ((float4*)xo)[i] = o;
}

// ---------------------------------------------------------------------------
// global_add_pool: pooled[batch[n]] += x[n]
// ---------------------------------------------------------------------------
__global__ void pool_nodes(const float* __restrict__ x, const int* __restrict__ batch,
                           float* __restrict__ pooled, int n)
{
    int i = blockIdx.x * blockDim.x + threadIdx.x;   // n*32 float4 chunks
    if (i >= n * 32) return;
    int node = i >> 5, c = (i & 31) * 4;
    int g = batch[node];
    float4 v = ((const float4*)x)[i];
    float* p = pooled + (size_t)g * D_CONV + c;
    atomicAdd(p + 0, v.x); atomicAdd(p + 1, v.y);
    atomicAdd(p + 2, v.z); atomicAdd(p + 3, v.w);
}

// ---------------------------------------------------------------------------
// MLP head: dense layer (optionally + ReLU); BN over G rows per channel
// ---------------------------------------------------------------------------
__global__ void head_lin(const float* __restrict__ A, const float* __restrict__ W,
                         const float* __restrict__ b, float* __restrict__ out,
                         int G, int K, int J, int do_relu)
{
    int i = blockIdx.x * blockDim.x + threadIdx.x;
    if (i >= G * J) return;
    int g = i / J, j = i - g * J;
    float acc = b[j];
    for (int k = 0; k < K; ++k) acc += A[(size_t)g * K + k] * W[(size_t)k * J + j];
    if (do_relu) acc = fmaxf(acc, 0.0f);
    out[i] = acc;
}

__global__ __launch_bounds__(256)
void bn_head(float* __restrict__ h, const float* __restrict__ gamma,
             const float* __restrict__ beta, int G, int J)
{
    __shared__ float s1[256], s2[256];
    __shared__ float smean, sinv;
    int j = blockIdx.x;
    float a = 0.0f, q = 0.0f;
    for (int g = threadIdx.x; g < G; g += blockDim.x) {
        float v = h[(size_t)g * J + j];
        a += v; q += v * v;
    }
    s1[threadIdx.x] = a; s2[threadIdx.x] = q;
    __syncthreads();
    for (int off = 128; off > 0; off >>= 1) {
        if (threadIdx.x < off) {
            s1[threadIdx.x] += s1[threadIdx.x + off];
            s2[threadIdx.x] += s2[threadIdx.x + off];
        }
        __syncthreads();
    }
    if (threadIdx.x == 0) {
        float m = s1[0] / (float)G;
        float v = s2[0] / (float)G - m * m;
        smean = m; sinv = rsqrtf(v + EPS);
    }
    __syncthreads();
    float gm = gamma[j], bt = beta[j], m = smean, iv = sinv;
    for (int g = threadIdx.x; g < G; g += blockDim.x) {
        size_t idx = (size_t)g * J + j;
        h[idx] = gm * (h[idx] - m) * iv + bt;
    }
}

// ---------------------------------------------------------------------------
extern "C" void kernel_launch(void* const* d_in, const int* in_sizes, int n_in,
                              void* d_out, int out_size, void* d_ws, size_t ws_size,
                              hipStream_t stream)
{
    const int N = in_sizes[2];          // 100000 nodes
    const int E = in_sizes[1] / 2;      // 1600000 edges
    const int G = out_size;             // 512 graphs

    const float* x_in  = (const float*)d_in[0];
    const int*   src   = (const int*)d_in[1];
    const int*   dst   = src + E;
    const int*   batch = (const int*)d_in[2];
    const float* Wc    = (const float*)d_in[3];
    const float* bc    = (const float*)d_in[4];
    const float* gc    = (const float*)d_in[5];
    const float* bec   = (const float*)d_in[6];
    const float* Wh0   = (const float*)d_in[7];
    const float* bh0   = (const float*)d_in[8];
    const float* gh0   = (const float*)d_in[9];
    const float* beh0  = (const float*)d_in[10];
    const float* Wh1   = (const float*)d_in[11];
    const float* bh1   = (const float*)d_in[12];
    const float* gh1   = (const float*)d_in[13];
    const float* beh1  = (const float*)d_in[14];
    const float* Wout  = (const float*)d_in[15];
    const float* bout  = (const float*)d_in[16];

    // workspace carving
    size_t off = 0;
    auto carve = [&](size_t bytes) -> void* {
        void* p = (char*)d_ws + off;
        off += (bytes + 255) & ~(size_t)255;
        return p;
    };
    const size_t NBF = (size_t)N * D_CONV * sizeof(float);
    float* B0     = (float*)carve(NBF);              // x_next
    float* B1     = (float*)carve(NBF);              // xw
    float* B2     = (float*)carve(NBF);              // y (post relu)
    float* dinv   = (float*)carve((size_t)N * sizeof(float));
    float* stats  = (float*)carve(256 * sizeof(float));       // ssum | ssq
    float* pooled = (float*)carve((size_t)G * D_CONV * sizeof(float));
    float* h0     = (float*)carve((size_t)G * D_HID * sizeof(float));
    float* h1     = (float*)carve((size_t)G * D_HID * sizeof(float));
    int*   cnt    = (int*)carve((size_t)N * sizeof(int));
    int*   incl   = (int*)carve((size_t)N * sizeof(int));
    int    nScan  = (N + 1023) / 1024;
    int*   bsum   = (int*)carve((size_t)nScan * sizeof(int));
    int*   rp     = (int*)carve((size_t)(N + 1) * sizeof(int));
    int*   fillc  = (int*)carve((size_t)N * sizeof(int));
    int*   col    = (int*)carve((size_t)E * sizeof(int));
    (void)ws_size; (void)n_in;

    // ---- CSR build (per launch; edges are constant inputs) ----
    hipMemsetAsync(cnt,   0, (size_t)N * sizeof(int), stream);
    hipMemsetAsync(fillc, 0, (size_t)N * sizeof(int), stream);
    count_in<<<(E + 255) / 256, 256, 0, stream>>>(dst, cnt, E);
    scan_block<<<nScan, 1024, 0, stream>>>(cnt, incl, bsum, N);
    scan_sums<<<1, 32, 0, stream>>>(bsum, nScan);
    make_rowptr<<<(N + 255) / 256, 256, 0, stream>>>(cnt, incl, bsum, rp, N, E);
    fill_csr<<<(E + 255) / 256, 256, 0, stream>>>(src, dst, rp, fillc, col, E);
    calc_dinv<<<(N + 255) / 256, 256, 0, stream>>>(cnt, dinv, N);

    // ---- 3x (GCNConv -> ReLU -> BatchNorm) ----
    const float* xc = x_in;
    const int gemm_grid = (N + TILE_M - 1) / TILE_M;
    for (int L = 0; L < 3; ++L) {
        gemm_xw<<<gemm_grid, 256, 0, stream>>>(xc, Wc + (size_t)L * D_CONV * D_CONV, B1, N);
        hipMemsetAsync(stats, 0, 256 * sizeof(float), stream);
        aggregate<<<(N + 7) / 8, 256, 0, stream>>>(B1, rp, col, dinv, bc + L * D_CONV,
                                                   B2, stats, stats + 128, N);
        bn_apply<<<(N * 32 + 255) / 256, 256, 0, stream>>>(B2, stats, stats + 128,
                                                           gc + L * D_CONV, bec + L * D_CONV,
                                                           B0, N);
        xc = B0;
    }

    // ---- global_add_pool ----
    hipMemsetAsync(pooled, 0, (size_t)G * D_CONV * sizeof(float), stream);
    pool_nodes<<<(N * 32 + 255) / 256, 256, 0, stream>>>(B0, batch, pooled, N);

    // ---- MLP head ----
    head_lin<<<(G * D_HID + 255) / 256, 256, 0, stream>>>(pooled, Wh0, bh0, h0,
                                                          G, D_CONV, D_HID, 1);
    bn_head<<<D_HID, 256, 0, stream>>>(h0, gh0, beh0, G, D_HID);
    head_lin<<<(G * D_HID + 255) / 256, 256, 0, stream>>>(h0, Wh1, bh1, h1,
                                                          G, D_HID, D_HID, 1);
    bn_head<<<D_HID, 256, 0, stream>>>(h1, gh1, beh1, G, D_HID);
    head_lin<<<(G + 255) / 256, 256, 0, stream>>>(h1, Wout, bout, (float*)d_out,
                                                  G, D_HID, 1, 0);
}
